// alignment_model_71287867179073
// MI455X (gfx1250) — compile-verified, matrix operand order checked by reading, not run on hardware
//
#include <hip/hip_runtime.h>
#include <math.h>

// ---------------------------------------------------------------------------
// CDNA5 (gfx1250) WMMA types and helpers
// ---------------------------------------------------------------------------
typedef __bf16 bf16x16 __attribute__((ext_vector_type(16)));
typedef float  f32x8   __attribute__((ext_vector_type(8)));

union FragB16 { bf16x16 v; uint4 q[2]; };
union AccF    { f32x8 v;  float f[8]; };

__device__ __forceinline__ float bf2f(unsigned short u) {
    return __uint_as_float(((unsigned)u) << 16);
}
__device__ __forceinline__ unsigned short f2bf(float f) {
    unsigned u = __float_as_uint(f);
    unsigned r = u + 0x7FFFu + ((u >> 16) & 1u);   // round-to-nearest-even
    return (unsigned short)(r >> 16);
}
__device__ __forceinline__ float sigm(float x) { return 1.f / (1.f + __expf(-x)); }

__device__ __forceinline__ void wmma_acc(AccF& acc, const FragB16& a, const FragB16& b) {
    // D(f32 16x16) = A(bf16 16x32) * B(bf16 32x16) + C
    acc.v = __builtin_amdgcn_wmma_f32_16x16x32_bf16(
        false, a.v, false, b.v, (short)0, acc.v, false, false);
}

// Load a 16x32 bf16 operand fragment from row-major [rows, ld] storage.
// ISA layout (16-bit A 16x32): lane L holds row (r0 + L%16); its 16 halves are
// K = k0 + (L/16)*8 + [0..7]  and  K = k0 + 16 + (L/16)*8 + [0..7].
__device__ __forceinline__ void load_frag(FragB16& f, const unsigned short* base,
                                          int ld, int r0, int k0, int lane) {
    int row  = r0 + (lane & 15);
    int koff = (lane >> 4) << 3;
    const unsigned short* p = base + row * ld + k0 + koff;
    f.q[0] = *(const uint4*)(p);
    f.q[1] = *(const uint4*)(p + 16);
}

// ---------------------------------------------------------------------------
// Generic WMMA GEMM:  C[M,N] = A[M,K](bf16) * W[N,K]^T(bf16)  (+ bias)
// block = 256 thr (8 waves); wave w -> N-tile n0 = bx*128 + w*16; m0 = by*16.
// The 16-row A panel is staged once per block into LDS with CDNA5 async
// global->LDS copies (ASYNCcnt), removing the 8x redundant A traffic; B
// streams from L2 directly into fragments. Requires K % 128 == 0.
// Cf (f32) and/or Cb (bf16) outputs, both with row stride ldc.
// ---------------------------------------------------------------------------
__global__ void k_gemm(const unsigned short* __restrict__ A, int lda,
                       const unsigned short* __restrict__ Bw, int ldb,
                       float* __restrict__ Cf, unsigned short* __restrict__ Cb,
                       int ldc, const float* __restrict__ bias,
                       int M, int N, int K) {
    __shared__ unsigned short shA[16 * 128];     // 4 KB A panel

    int tid = threadIdx.x;
    int wave = tid >> 5, lane = tid & 31;
    int n0 = blockIdx.x * 128 + wave * 16;
    int m0 = blockIdx.y * 16;
    int koff = (lane >> 4) << 3;
    int nrow = n0 + (lane & 15); if (nrow >= N) nrow = N - 1;   // clamp pad tiles
    const unsigned short* bp = Bw + (size_t)nrow * ldb + koff;

    // per-thread async-copy unit: row cr (0..15), 8-half chunk cc (16B)
    int cr = tid >> 4;
    int cc = (tid & 15) << 3;
    const unsigned short* asrc = A + (size_t)(m0 + cr) * lda + cc;
    unsigned ldst = (unsigned)(uintptr_t)&shA[cr * 128 + cc];

    AccF acc;
    #pragma unroll
    for (int r = 0; r < 8; r++) acc.f[r] = 0.f;

    for (int kp = 0; kp < K; kp += 128) {
        // ---- async stage A[m0..m0+15, kp..kp+128) -> LDS ----
        {
            unsigned long long ga = (unsigned long long)(uintptr_t)(asrc + kp);
            asm volatile("global_load_async_to_lds_b128 %0, %1, off"
                         :: "v"(ldst), "v"(ga) : "memory");
            asm volatile("s_wait_asynccnt 0" ::: "memory");
        }
        __syncthreads();
        if (kp + 128 < K)
            __builtin_prefetch(bp + kp + 128, 0, 1);   // next B panel -> global_prefetch
        #pragma unroll
        for (int kc = 0; kc < 4; kc++) {
            FragB16 a, b;
            const unsigned short* p = shA + (lane & 15) * 128 + kc * 32 + koff;
            a.q[0] = *(const uint4*)(p); a.q[1] = *(const uint4*)(p + 16);
            const unsigned short* q = bp + kp + kc * 32;
            b.q[0] = *(const uint4*)(q); b.q[1] = *(const uint4*)(q + 16);
            wmma_acc(acc, a, b);
        }
        __syncthreads();
    }

    // f32 C/D layout: VGPR r -> M = r + (lane<16 ? 0 : 8), N = lane%16
    int mb = m0 + ((lane >> 4) << 3);
    int n  = n0 + (lane & 15);
    if (n < N) {
        float bv = bias ? bias[n] : 0.f;
        #pragma unroll
        for (int r = 0; r < 8; r++) {
            int m = mb + r;
            if (m < M) {
                float v = acc.f[r] + bv;
                if (Cf) Cf[(size_t)m * ldc + n] = v;
                if (Cb) Cb[(size_t)m * ldc + n] = f2bf(v);
            }
        }
    }
}

// ---------------------------------------------------------------------------
// Conv1d(K=9,pad=4) as implicit-im2col WMMA GEMM, fused BN(affine)+ReLU.
// X: bf16 [(b*1024+t), Cin]; Wr: bf16 [Cout, 9*Cin] with col = kk*Cin+ci.
// Cin = 1<<cs is a multiple of 32 so each 32-wide K-chunk stays in one tap kk.
// ---------------------------------------------------------------------------
__global__ void k_conv_gemm(const unsigned short* __restrict__ X,
                            const unsigned short* __restrict__ Wr, int cs,
                            const float* __restrict__ cb, const float* __restrict__ bg,
                            const float* __restrict__ bb,
                            unsigned short* __restrict__ Y, int Cout) {
    int wave = threadIdx.x >> 5, lane = threadIdx.x & 31;
    int n0 = blockIdx.x * 128 + wave * 16;
    int m0 = blockIdx.y * 16;
    int koff = (lane >> 4) << 3;
    int Cin = 1 << cs, K = 9 << cs;

    int m  = m0 + (lane & 15);
    int b_ = m >> 10, t_ = m & 1023;
    const unsigned short* bp = Wr + (size_t)(n0 + (lane & 15)) * K + koff;

    AccF acc;
    #pragma unroll
    for (int r = 0; r < 8; r++) acc.f[r] = 0.f;

    for (int k0 = 0; k0 < K; k0 += 32) {
        int kk = k0 >> cs, ci0 = k0 - (kk << cs);
        int ts = t_ + kk - 4;
        FragB16 a, b;
        if ((unsigned)ts < 1024u) {
            const unsigned short* p = X + (size_t)(((b_ << 10) + ts) << cs) + ci0 + koff;
            a.q[0] = *(const uint4*)(p); a.q[1] = *(const uint4*)(p + 16);
        } else {
            a.q[0] = make_uint4(0u, 0u, 0u, 0u); a.q[1] = make_uint4(0u, 0u, 0u, 0u);
        }
        b.q[0] = *(const uint4*)(bp + k0); b.q[1] = *(const uint4*)(bp + k0 + 16);
        wmma_acc(acc, a, b);
    }

    int mb = m0 + ((lane >> 4) << 3);
    int n  = n0 + (lane & 15);
    float s = bg[n], cbn = cb[n], sh = bb[n];
    #pragma unroll
    for (int r = 0; r < 8; r++) {
        int mm = mb + r;
        float v = s * (acc.f[r] + cbn) + sh;
        v = fmaxf(v, 0.f);
        Y[(size_t)mm * Cout + n] = f2bf(v);
    }
}

// ---------------------------------------------------------------------------
// Encoder bi-LSTM recurrence: persistent, one workgroup per direction.
// grid=2 (dir), block=512 (16 waves). h/c/z live in LDS (56KB of 320KB WGP LDS).
// z[16,1024] = h[16,256] @ Whh^T (64 WMMA tiles/step) + Zx(precomputed) + b.
// ---------------------------------------------------------------------------
__global__ void k_enc_lstm(const unsigned short* __restrict__ whh_fw,
                           const unsigned short* __restrict__ whh_bw,
                           const unsigned short* __restrict__ zx_fw,
                           const unsigned short* __restrict__ zx_bw,
                           const float* __restrict__ b_fw,
                           const float* __restrict__ b_bw,
                           unsigned short* __restrict__ hsrc) {
    int dir = blockIdx.x;
    const unsigned short* whh = dir ? whh_bw : whh_fw;
    const unsigned short* zx  = dir ? zx_bw  : zx_fw;
    const float* bias         = dir ? b_bw   : b_fw;

    __shared__ unsigned short sh_h[16 * 256];   // 8 KB   h (bf16)
    __shared__ unsigned short sh_z[16 * 1024];  // 32 KB  gate pre-acts (bf16)
    __shared__ float          sh_c[16 * 256];   // 16 KB  cell state

    int tid = threadIdx.x;
    for (int i = tid; i < 16 * 256; i += 512) { sh_h[i] = 0; sh_c[i] = 0.f; }
    __syncthreads();

    int wave = tid >> 5, lane = tid & 31;
    int koff = (lane >> 4) << 3;

    for (int step = 0; step < 1024; step++) {
        int t = dir ? (1023 - step) : step;

        AccF acc[4];
        #pragma unroll
        for (int j = 0; j < 4; j++)
            #pragma unroll
            for (int r = 0; r < 8; r++) acc[j].f[r] = 0.f;

        #pragma unroll
        for (int kc = 0; kc < 8; kc++) {
            FragB16 a;
            const unsigned short* p = sh_h + (lane & 15) * 256 + kc * 32 + koff;
            a.q[0] = *(const uint4*)(p); a.q[1] = *(const uint4*)(p + 16);
            #pragma unroll
            for (int j = 0; j < 4; j++) {
                int n0 = (wave * 4 + j) * 16;
                FragB16 b;
                const unsigned short* q = whh + (size_t)(n0 + (lane & 15)) * 256 + kc * 32 + koff;
                b.q[0] = *(const uint4*)(q); b.q[1] = *(const uint4*)(q + 16);
                wmma_acc(acc[j], a, b);
            }
        }

        int mb = (lane >= 16) ? 8 : 0;
        #pragma unroll
        for (int j = 0; j < 4; j++) {
            int n = (wave * 4 + j) * 16 + (lane & 15);
            float bv = bias[n];
            #pragma unroll
            for (int r = 0; r < 8; r++) {
                int m = mb + r;
                float v = acc[j].f[r] + bv + bf2f(zx[(size_t)((m << 10) + t) * 1024 + n]);
                sh_z[m * 1024 + n] = f2bf(v);
            }
        }
        __syncthreads();

        #pragma unroll
        for (int e = 0; e < 8; e++) {
            int idx = tid + e * 512;
            int b_ = idx >> 8, u = idx & 255;
            float iG = bf2f(sh_z[b_ * 1024 + u]);
            float fG = bf2f(sh_z[b_ * 1024 + 256 + u]);
            float gG = bf2f(sh_z[b_ * 1024 + 512 + u]);
            float oG = bf2f(sh_z[b_ * 1024 + 768 + u]);
            float c = sigm(fG) * sh_c[b_ * 256 + u] + sigm(iG) * tanhf(gG);
            sh_c[b_ * 256 + u] = c;
            float h = sigm(oG) * tanhf(c);
            unsigned short hb = f2bf(h);
            sh_h[b_ * 256 + u] = hb;
            hsrc[(size_t)((b_ << 10) + t) * 512 + (dir << 8) + u] = hb;
        }
        __syncthreads();
    }
}

// ---------------------------------------------------------------------------
// Decoder LSTM step (shared by layer 1 and 2): z = Acat[16,2048] @ Wcat^T + b.
// grid=16 blocks; block j owns h columns [j*64, j*64+64) and computes all 4
// gate column groups for its slice (no cross-block sync). 8 waves x 2 tiles.
// ---------------------------------------------------------------------------
__global__ void k_dec_lstm(const unsigned short* __restrict__ Acat,
                           const unsigned short* __restrict__ Wcat,
                           const float* __restrict__ bias,
                           float* __restrict__ cst,
                           unsigned short* __restrict__ h_out) {
    int j = blockIdx.x;
    __shared__ float sh_z[16 * 256];   // 16 KB: local cols = gate*64 + ul

    int tid = threadIdx.x;
    int wave = tid >> 5, lane = tid & 31;
    int koff = (lane >> 4) << 3;
    const unsigned short* ap = Acat + (size_t)(lane & 15) * 2048 + koff;

    AccF acc[2];
    const unsigned short* bp[2];
    #pragma unroll
    for (int jj = 0; jj < 2; jj++) {
        int tl = wave * 2 + jj;
        int n0 = ((tl >> 2) << 10) + j * 64 + ((tl & 3) << 4);
        bp[jj] = Wcat + (size_t)(n0 + (lane & 15)) * 2048 + koff;
        #pragma unroll
        for (int r = 0; r < 8; r++) acc[jj].f[r] = 0.f;
    }

    for (int k0 = 0; k0 < 2048; k0 += 32) {
        FragB16 a;
        a.q[0] = *(const uint4*)(ap + k0); a.q[1] = *(const uint4*)(ap + k0 + 16);
        #pragma unroll
        for (int jj = 0; jj < 2; jj++) {
            FragB16 b;
            b.q[0] = *(const uint4*)(bp[jj] + k0); b.q[1] = *(const uint4*)(bp[jj] + k0 + 16);
            wmma_acc(acc[jj], a, b);
        }
    }

    int mb = (lane >= 16) ? 8 : 0;
    #pragma unroll
    for (int jj = 0; jj < 2; jj++) {
        int tl = wave * 2 + jj, gate = tl >> 2, ncol = tl & 3;
        int lc = gate * 64 + ncol * 16 + (lane & 15);
        int n  = (gate << 10) + j * 64 + ncol * 16 + (lane & 15);
        float bv = bias[n];
        #pragma unroll
        for (int r = 0; r < 8; r++) sh_z[(mb + r) * 256 + lc] = acc[jj].f[r] + bv;
    }
    __syncthreads();

    #pragma unroll
    for (int e = 0; e < 4; e++) {
        int idx = tid + e * 256;
        int b_ = idx >> 6, ul = idx & 63;
        int u = j * 64 + ul;
        float iG = sh_z[b_ * 256 + ul];
        float fG = sh_z[b_ * 256 + 64 + ul];
        float gG = sh_z[b_ * 256 + 128 + ul];
        float oG = sh_z[b_ * 256 + 192 + ul];
        float c = sigm(fG) * cst[b_ * 1024 + u] + sigm(iG) * tanhf(gG);
        cst[b_ * 1024 + u] = c;
        h_out[b_ * 1024 + u] = f2bf(sigm(oG) * tanhf(c));
    }
}

// ---------------------------------------------------------------------------
// Attention / elementwise kernels
// ---------------------------------------------------------------------------
// loc[n,l,tm] = sum_k F[l,k] * cum[n, tm+k-15]   (K=31, pad 15) -> bf16 [n*1024+tm, 128]
__global__ void k_loc_conv(const float* __restrict__ cum, const float* __restrict__ F,
                           unsigned short* __restrict__ loc) {
    int bid = blockIdx.x;            // n*1024 + tm
    int n = bid >> 10, tm = bid & 1023;
    __shared__ float sh[31];
    int tid = threadIdx.x;           // 128
    if (tid < 31) {
        int ts = tm + tid - 15;
        sh[tid] = ((unsigned)ts < 1024u) ? cum[n * 1024 + ts] : 0.f;
    }
    __syncthreads();
    float s = 0.f;
    #pragma unroll
    for (int k = 0; k < 31; k++) s += F[tid * 31 + k] * sh[k];
    loc[(size_t)bid * 128 + tid] = f2bf(s);
}

// e[n,tm] = sum_a v[a] * tanh(Ws[n,a] + Vh[n,tm,a] + Uf[n,tm,a])
__global__ void k_energy(const float* __restrict__ Ws, const float* __restrict__ Vh,
                         const float* __restrict__ Uf, const float* __restrict__ vv,
                         float* __restrict__ e) {
    int tm = blockIdx.x, n = blockIdx.y, a = threadIdx.x;   // 256
    size_t row = (size_t)(n * 1024 + tm) * 256;
    float p = vv[a] * tanhf(Ws[n * 256 + a] + Vh[row + a] + Uf[row + a]);
    __shared__ float red[256];
    red[a] = p; __syncthreads();
    for (int s = 128; s > 0; s >>= 1) {
        if (a < s) red[a] += red[a + s];
        __syncthreads();
    }
    if (a == 0) e[n * 1024 + tm] = red[0];
}

// softmax over tm, cum += att, ctx[n,:512] = att @ h_src.  grid=16, block=1024.
__global__ void k_softmax_ctx(const float* __restrict__ e, float* __restrict__ cum,
                              const unsigned short* __restrict__ hsrc,
                              unsigned short* __restrict__ ctx_bf) {
    int n = blockIdx.x, tid = threadIdx.x;
    __shared__ float red[1024];
    __shared__ float att[1024];
    float x = e[n * 1024 + tid];
    red[tid] = x; __syncthreads();
    for (int s = 512; s > 0; s >>= 1) {
        if (tid < s) red[tid] = fmaxf(red[tid], red[tid + s]);
        __syncthreads();
    }
    float mx = red[0]; __syncthreads();
    float ex = __expf(x - mx);
    red[tid] = ex; __syncthreads();
    for (int s = 512; s > 0; s >>= 1) {
        if (tid < s) red[tid] += red[tid + s];
        __syncthreads();
    }
    float a = ex / red[0];
    att[tid] = a;
    cum[n * 1024 + tid] += a;
    __syncthreads();
    if (tid < 512) {
        float s = 0.f;
        for (int tm = 0; tm < 1024; tm++)
            s += att[tm] * bf2f(hsrc[(size_t)(n * 1024 + tm) * 512 + tid]);
        ctx_bf[n * 512 + tid] = f2bf(s);
    }
}

// xcat1 = [ emb[ipa[n,t]] | ctx | h1_prev ]   (16 x 2048 bf16)
__global__ void k_prep1(const float* __restrict__ emb, const int* __restrict__ ipa, int t,
                        const unsigned short* __restrict__ ctx_bf,
                        const unsigned short* __restrict__ h1_bf,
                        unsigned short* __restrict__ xcat) {
    int idx = blockIdx.x * 1024 + threadIdx.x;     // 16*2048
    int b_ = idx >> 11, c = idx & 2047;
    unsigned short v;
    if (c < 512)        v = f2bf(emb[(size_t)ipa[b_ * 64 + t] * 512 + c]);
    else if (c < 1024)  v = ctx_bf[b_ * 512 + (c - 512)];
    else                v = h1_bf[b_ * 1024 + (c - 1024)];
    xcat[idx] = v;
}

// xcat2 = [ h1 | h2_prev ]
__global__ void k_prep2(const unsigned short* __restrict__ h1_bf,
                        const unsigned short* __restrict__ h2_bf,
                        unsigned short* __restrict__ xcat) {
    int idx = blockIdx.x * 1024 + threadIdx.x;
    int b_ = idx >> 11, c = idx & 2047;
    xcat[idx] = (c < 1024) ? h1_bf[b_ * 1024 + c] : h2_bf[b_ * 1024 + (c - 1024)];
}

// acat = [ h2 | ctx ]  (16 x 1536)
__global__ void k_prepcc(const unsigned short* __restrict__ h2_bf,
                         const unsigned short* __restrict__ ctx_bf,
                         unsigned short* __restrict__ acat) {
    int idx = blockIdx.x * 1024 + threadIdx.x;     // 16*1536 = 24576
    int b_ = idx / 1536, c = idx % 1536;
    acat[idx] = (c < 1024) ? h2_bf[b_ * 1024 + c] : ctx_bf[b_ * 512 + (c - 1024)];
}

// log_softmax over 100 vocab cols.  grid = 1024 rows, block = 128.
__global__ void k_logsoftmax(const float* __restrict__ logits, float* __restrict__ out) {
    int row = blockIdx.x, tid = threadIdx.x;
    __shared__ float red[128];
    float x = (tid < 100) ? logits[(size_t)row * 100 + tid] : -1e30f;
    red[tid] = x; __syncthreads();
    for (int s = 64; s > 0; s >>= 1) {
        if (tid < s) red[tid] = fmaxf(red[tid], red[tid + s]);
        __syncthreads();
    }
    float mx = red[0]; __syncthreads();
    float ex = (tid < 100) ? __expf(x - mx) : 0.f;
    red[tid] = ex; __syncthreads();
    for (int s = 64; s > 0; s >>= 1) {
        if (tid < s) red[tid] += red[tid + s];
        __syncthreads();
    }
    if (tid < 100) out[(size_t)row * 100 + tid] = x - mx - logf(red[0]);
}

// ---------------------------------------------------------------------------
// Conversions / layout kernels
// ---------------------------------------------------------------------------
__global__ void k_cvt(const float* __restrict__ s, unsigned short* __restrict__ d, int n) {
    for (int i = blockIdx.x * 256 + threadIdx.x; i < n; i += gridDim.x * 256)
        d[i] = f2bf(s[i]);
}
// conv weight [Cout, Cin, 9] -> bf16 [Cout, 9*Cin] with col = kk*Cin + ci
__global__ void k_cvt_convw(const float* __restrict__ s, unsigned short* __restrict__ d,
                            int Cin, int n) {
    for (int i = blockIdx.x * 256 + threadIdx.x; i < n; i += gridDim.x * 256) {
        int co = i / (Cin * 9), r = i % (Cin * 9);
        int ci = r / 9, kk = r % 9;
        d[(size_t)co * 9 * Cin + kk * Cin + ci] = f2bf(s[i]);
    }
}
// concat [R,ca] | [R,cb] -> bf16 [R, ca+cb]
__global__ void k_catw(const float* __restrict__ a, const float* __restrict__ b,
                       unsigned short* __restrict__ d, int ca, int cb, int n) {
    int w = ca + cb;
    for (int i = blockIdx.x * 256 + threadIdx.x; i < n; i += gridDim.x * 256) {
        int r = i / w, c = i % w;
        d[i] = f2bf(c < ca ? a[(size_t)r * ca + c] : b[(size_t)r * cb + (c - ca)]);
    }
}
// mel [16,128,1024] -> bf16 X0[(b*1024+t)*128 + c]
__global__ void k_mel(const float* __restrict__ mel, unsigned short* __restrict__ X0, int n) {
    for (int i = blockIdx.x * 256 + threadIdx.x; i < n; i += gridDim.x * 256) {
        int b_ = i >> 17, rem = i & 131071;
        int t = rem >> 7, c = rem & 127;
        X0[i] = f2bf(mel[(size_t)((b_ << 7) + c) * 1024 + t]);
    }
}
__global__ void k_zero32(unsigned int* __restrict__ p, int n) {
    for (int i = blockIdx.x * 256 + threadIdx.x; i < n; i += gridDim.x * 256) p[i] = 0u;
}

// ---------------------------------------------------------------------------
// Host orchestration
// ---------------------------------------------------------------------------
extern "C" void kernel_launch(void* const* d_in, const int* in_sizes, int n_in,
                              void* d_out, int out_size, void* d_ws, size_t ws_size,
                              hipStream_t stream) {
    (void)in_sizes; (void)n_in; (void)out_size; (void)ws_size;

    const float* mel       = (const float*)d_in[0];
    const int*   ipa       = (const int*)d_in[1];
    const float* conv1_w   = (const float*)d_in[2];
    const float* conv1_b   = (const float*)d_in[3];
    const float* bn1_g     = (const float*)d_in[4];
    const float* bn1_b     = (const float*)d_in[5];
    const float* conv2_w   = (const float*)d_in[6];
    const float* conv2_b   = (const float*)d_in[7];
    const float* bn2_g     = (const float*)d_in[8];
    const float* bn2_b     = (const float*)d_in[9];
    const float* conv3_w   = (const float*)d_in[10];
    const float* conv3_b   = (const float*)d_in[11];
    const float* bn3_g     = (const float*)d_in[12];
    const float* bn3_b     = (const float*)d_in[13];
    const float* enc_fw_wih = (const float*)d_in[14];
    const float* enc_fw_whh = (const float*)d_in[15];
    const float* enc_fw_b   = (const float*)d_in[16];
    const float* enc_bw_wih = (const float*)d_in[17];
    const float* enc_bw_whh = (const float*)d_in[18];
    const float* enc_bw_b   = (const float*)d_in[19];
    const float* att_F     = (const float*)d_in[20];
    const float* att_W     = (const float*)d_in[21];
    const float* att_V     = (const float*)d_in[22];
    const float* att_U     = (const float*)d_in[23];
    const float* att_v     = (const float*)d_in[24];
    const float* emb       = (const float*)d_in[25];
    const float* dec1_wih  = (const float*)d_in[26];
    const float* dec1_whh  = (const float*)d_in[27];
    const float* dec1_b    = (const float*)d_in[28];
    const float* dec2_wih  = (const float*)d_in[29];
    const float* dec2_whh  = (const float*)d_in[30];
    const float* dec2_b    = (const float*)d_in[31];
    const float* cc_w      = (const float*)d_in[32];
    const float* cc_b      = (const float*)d_in[33];
    const float* gen_w     = (const float*)d_in[34];
    const float* gen_b     = (const float*)d_in[35];

    const int BT = 16 * 1024;

    // workspace bump allocator (deterministic per-call layout)
    char* wp = (char*)d_ws;
    auto alloc = [&](size_t bytes) -> void* {
        void* r = (void*)wp;
        wp += (bytes + 255) & ~(size_t)255;
        return r;
    };
    unsigned short* X0   = (unsigned short*)alloc((size_t)BT * 128 * 2);
    unsigned short* W1r  = (unsigned short*)alloc((size_t)512 * 1152 * 2);
    unsigned short* W2r  = (unsigned short*)alloc((size_t)512 * 4608 * 2);
    unsigned short* W3r  = (unsigned short*)alloc((size_t)512 * 4608 * 2);
    unsigned short* Y1   = (unsigned short*)alloc((size_t)BT * 512 * 2);
    unsigned short* Y2   = (unsigned short*)alloc((size_t)BT * 512 * 2);
    unsigned short* Y3   = (unsigned short*)alloc((size_t)BT * 512 * 2);
    unsigned short* wihF = (unsigned short*)alloc((size_t)1024 * 512 * 2);
    unsigned short* wihB = (unsigned short*)alloc((size_t)1024 * 512 * 2);
    unsigned short* whhF = (unsigned short*)alloc((size_t)1024 * 256 * 2);
    unsigned short* whhB = (unsigned short*)alloc((size_t)1024 * 256 * 2);
    unsigned short* zxF  = (unsigned short*)alloc((size_t)BT * 1024 * 2);
    unsigned short* zxB  = (unsigned short*)alloc((size_t)BT * 1024 * 2);
    unsigned short* hsrc = (unsigned short*)alloc((size_t)BT * 512 * 2);
    float*          Vh   = (float*)alloc((size_t)BT * 256 * 4);
    unsigned short* aWb  = (unsigned short*)alloc((size_t)256 * 1024 * 2);
    unsigned short* aVb  = (unsigned short*)alloc((size_t)256 * 512 * 2);
    unsigned short* aUb  = (unsigned short*)alloc((size_t)256 * 128 * 2);
    unsigned short* Wc1  = (unsigned short*)alloc((size_t)4096 * 2048 * 2);
    unsigned short* Wc2  = (unsigned short*)alloc((size_t)4096 * 2048 * 2);
    unsigned short* ccWb = (unsigned short*)alloc((size_t)1024 * 1536 * 2);
    unsigned short* genWb= (unsigned short*)alloc((size_t)100 * 1024 * 2);
    unsigned short* locB = (unsigned short*)alloc((size_t)BT * 128 * 2);
    float*          Uf   = (float*)alloc((size_t)BT * 256 * 4);
    float*          Ws   = (float*)alloc((size_t)16 * 256 * 4);
    float*          eB   = (float*)alloc((size_t)16 * 1024 * 4);
    float*          cum  = (float*)alloc((size_t)16 * 1024 * 4);
    unsigned short* ctxB = (unsigned short*)alloc((size_t)16 * 512 * 2);
    unsigned short* xc1  = (unsigned short*)alloc((size_t)16 * 2048 * 2);
    unsigned short* xc2  = (unsigned short*)alloc((size_t)16 * 2048 * 2);
    unsigned short* acc_ = (unsigned short*)alloc((size_t)16 * 1536 * 2);
    unsigned short* h1B  = (unsigned short*)alloc((size_t)16 * 1024 * 2);
    unsigned short* h2B  = (unsigned short*)alloc((size_t)16 * 1024 * 2);
    float*          c1   = (float*)alloc((size_t)16 * 1024 * 4);
    float*          c2   = (float*)alloc((size_t)16 * 1024 * 4);
    float*          hT   = (float*)alloc((size_t)1024 * 1024 * 4);
    unsigned short* hTb  = (unsigned short*)alloc((size_t)1024 * 1024 * 2);
    float*          lg   = (float*)alloc((size_t)1024 * 100 * 4);

    auto cvt  = [&](const float* s, unsigned short* d, int n) {
        k_cvt<<<dim3((n + 255) / 256), dim3(256), 0, stream>>>(s, d, n);
    };
    auto gemm = [&](const unsigned short* A, int lda, const unsigned short* B, int ldb,
                    float* Cf, unsigned short* Cb, int ldc, const float* bias,
                    int M, int N, int K) {
        dim3 g((N + 127) / 128, M / 16);
        k_gemm<<<g, dim3(256), 0, stream>>>(A, lda, B, ldb, Cf, Cb, ldc, bias, M, N, K);
    };

    // ---- state zero-init (poisoned workspace) ----
    k_zero32<<<dim3(64), dim3(256), 0, stream>>>((unsigned int*)cum, 16384);
    k_zero32<<<dim3(64), dim3(256), 0, stream>>>((unsigned int*)c1, 16384);
    k_zero32<<<dim3(64), dim3(256), 0, stream>>>((unsigned int*)c2, 16384);
    k_zero32<<<dim3(32), dim3(256), 0, stream>>>((unsigned int*)h1B, 8192);
    k_zero32<<<dim3(32), dim3(256), 0, stream>>>((unsigned int*)h2B, 8192);

    // ---- bf16 weight/activation staging ----
    k_mel<<<dim3(4096), dim3(256), 0, stream>>>(mel, X0, BT * 128);
    k_cvt_convw<<<dim3(2048), dim3(256), 0, stream>>>(conv1_w, W1r, 128, 512 * 128 * 9);
    k_cvt_convw<<<dim3(4096), dim3(256), 0, stream>>>(conv2_w, W2r, 512, 512 * 512 * 9);
    k_cvt_convw<<<dim3(4096), dim3(256), 0, stream>>>(conv3_w, W3r, 512, 512 * 512 * 9);
    cvt(enc_fw_wih, wihF, 1024 * 512);  cvt(enc_bw_wih, wihB, 1024 * 512);
    cvt(enc_fw_whh, whhF, 1024 * 256);  cvt(enc_bw_whh, whhB, 1024 * 256);
    cvt(att_W, aWb, 256 * 1024);  cvt(att_V, aVb, 256 * 512);  cvt(att_U, aUb, 256 * 128);
    k_catw<<<dim3(8192), dim3(256), 0, stream>>>(dec1_wih, dec1_whh, Wc1, 1024, 1024, 4096 * 2048);
    k_catw<<<dim3(8192), dim3(256), 0, stream>>>(dec2_wih, dec2_whh, Wc2, 1024, 1024, 4096 * 2048);
    cvt(cc_w, ccWb, 1024 * 1536);  cvt(gen_w, genWb, 100 * 1024);

    // ---- prenet: 3x (conv1d K=9 -> BN affine -> relu) as implicit-im2col WMMA GEMM ----
    k_conv_gemm<<<dim3(4, 1024), dim3(256), 0, stream>>>(X0, W1r, 7, conv1_b, bn1_g, bn1_b, Y1, 512);
    k_conv_gemm<<<dim3(4, 1024), dim3(256), 0, stream>>>(Y1, W2r, 9, conv2_b, bn2_g, bn2_b, Y2, 512);
    k_conv_gemm<<<dim3(4, 1024), dim3(256), 0, stream>>>(Y2, W3r, 9, conv3_b, bn3_g, bn3_b, Y3, 512);

    // ---- encoder: time-parallel x@Wih^T, then persistent LDS recurrence ----
    gemm(Y3, 512, wihF, 512, nullptr, zxF, 1024, nullptr, BT, 1024, 512);
    gemm(Y3, 512, wihB, 512, nullptr, zxB, 1024, nullptr, BT, 1024, 512);
    k_enc_lstm<<<dim3(2), dim3(512), 0, stream>>>(whhF, whhB, zxF, zxB, enc_fw_b, enc_bw_b, hsrc);

    // ---- attention precompute Vh = h_src @ V^T ----
    gemm(hsrc, 512, aVb, 512, Vh, nullptr, 256, nullptr, BT, 256, 512);

    // ---- decoder: 64 steps ----
    for (int t = 0; t < 64; t++) {
        gemm(h2B, 1024, aWb, 1024, Ws, nullptr, 256, nullptr, 16, 256, 1024);          // Ws
        k_loc_conv<<<dim3(BT), dim3(128), 0, stream>>>(cum, att_F, locB);              // loc feats
        gemm(locB, 128, aUb, 128, Uf, nullptr, 256, nullptr, BT, 256, 128);            // Uf
        k_energy<<<dim3(1024, 16), dim3(256), 0, stream>>>(Ws, Vh, Uf, att_v, eB);     // energies
        k_softmax_ctx<<<dim3(16), dim3(1024), 0, stream>>>(eB, cum, hsrc, ctxB);       // att,ctx
        k_prep1<<<dim3(32), dim3(1024), 0, stream>>>(emb, ipa, t, ctxB, h1B, xc1);
        k_dec_lstm<<<dim3(16), dim3(256), 0, stream>>>(xc1, Wc1, dec1_b, c1, h1B);
        k_prep2<<<dim3(32), dim3(1024), 0, stream>>>(h1B, h2B, xc2);
        k_dec_lstm<<<dim3(16), dim3(256), 0, stream>>>(xc2, Wc2, dec2_b, c2, h2B);
        k_prepcc<<<dim3(24), dim3(1024), 0, stream>>>(h2B, ctxB, acc_);
        // h_tilde rows are (n*64 + t): m-stride 64*1024, base offset t*1024
        dim3 gcc((1024 + 127) / 128, 1);
        k_gemm<<<gcc, dim3(256), 0, stream>>>(acc_, 1536, ccWb, 1536,
                                              hT + (size_t)t * 1024, hTb + (size_t)t * 1024,
                                              64 * 1024, cc_b, 16, 1024, 1536);
    }

    // ---- generator + log_softmax ----
    gemm(hTb, 1024, genWb, 1024, lg, nullptr, 100, gen_b, 1024, 100, 1024);
    k_logsoftmax<<<dim3(1024), dim3(128), 0, stream>>>(lg, (float*)d_out);
}
